// SparseMoEBlock_12412455485618
// MI455X (gfx1250) — matrix operand order repaired and checked by reading, not compiled, not run on hardware
//
#include <hip/hip_runtime.h>
#include <cstdint>

// ---------------- problem constants (from reference) ----------------
constexpr int Tn = 2048;   // tokens = B*S
constexpr int HD = 1024;   // hidden
constexpr int ID = 2816;   // intermediate
constexpr int NE = 8;      // experts
constexpr int BM = 32;     // tile M
constexpr int BN = 128;    // tile N (each of 4 n-waves owns 2 16-wide subtiles)
constexpr int BK = 64;     // K chunk per staging round (2 WMMA K-steps)
constexpr int APAD = 8;    // row pad (u16) -> 144B stride: 16B-aligned, 16 distinct banks

typedef __attribute__((ext_vector_type(16))) __bf16 v16bf;
typedef __attribute__((ext_vector_type(8)))  float  v8f;
typedef unsigned int   u32;
typedef unsigned short u16;
typedef __attribute__((ext_vector_type(4))) u32   u32x4;
typedef __attribute__((ext_vector_type(2))) u32   u32x2;
typedef __attribute__((ext_vector_type(4))) float f32x4;

union FragU { v16bf v; u32 u[8]; };

__device__ __forceinline__ u16 f2bf(float f) {
  u32 u = __builtin_bit_cast(u32, f);
  u += 0x7FFFu + ((u >> 16) & 1u);   // round-to-nearest-even
  return (u16)(u >> 16);
}

// ---------------- kernel 1: fp32 -> bf16 activation convert (vectorized) ---
__global__ void k_cvt_bf16(const float* __restrict__ x, u16* __restrict__ xb) {
  int i = blockIdx.x * 256 + threadIdx.x;          // one float4 per thread
  f32x4 v = ((const f32x4*)x)[i];
  u32x2 o;
  o.x = (u32)f2bf(v.x) | ((u32)f2bf(v.y) << 16);
  o.y = (u32)f2bf(v.z) | ((u32)f2bf(v.w) << 16);
  ((u32x2*)xb)[i] = o;
}

// ---------------- kernel 2: router (logits, softmax, top-2, renorm) --------
__global__ void __launch_bounds__(256) k_router(
    const float* __restrict__ x, const float* __restrict__ gw,
    float* __restrict__ logits_out, int* __restrict__ topk_id,
    float* __restrict__ topk_w) {
  int wv   = threadIdx.x >> 5;
  int lane = threadIdx.x & 31;
  int t    = blockIdx.x * 8 + wv;
  if (t >= Tn) return;

  float acc[NE];
#pragma unroll
  for (int e = 0; e < NE; ++e) acc[e] = 0.f;

  const float* xr = x + (size_t)t * HD;
  for (int h = lane; h < HD; h += 32) {
    float xv = xr[h];
#pragma unroll
    for (int e = 0; e < NE; ++e) acc[e] += xv * gw[e * HD + h];
  }
#pragma unroll
  for (int off = 16; off > 0; off >>= 1) {
#pragma unroll
    for (int e = 0; e < NE; ++e) acc[e] += __shfl_xor(acc[e], off, 32);
  }

  if (lane == 0) {
    float mx = acc[0];
#pragma unroll
    for (int e = 1; e < NE; ++e) mx = fmaxf(mx, acc[e]);
    float p[NE], s = 0.f;
#pragma unroll
    for (int e = 0; e < NE; ++e) { p[e] = __expf(acc[e] - mx); s += p[e]; }
    float inv = 1.f / s;
#pragma unroll
    for (int e = 0; e < NE; ++e) {
      p[e] *= inv;
      logits_out[(size_t)t * NE + e] = acc[e];   // raw logits (reference output #2)
    }
    int i1 = 0;
#pragma unroll
    for (int e = 1; e < NE; ++e) if (p[e] > p[i1]) i1 = e;
    int i2 = (i1 == 0) ? 1 : 0;
#pragma unroll
    for (int e = 0; e < NE; ++e) if (e != i1 && p[e] > p[i2]) i2 = e;
    float rn = 1.f / (p[i1] + p[i2]);
    topk_id[2 * t]     = i1;  topk_w[2 * t]     = p[i1] * rn;
    topk_id[2 * t + 1] = i2;  topk_w[2 * t + 1] = p[i2] * rn;
  }
}

// ---------------- kernel 3: deterministic per-expert token lists -----------
__global__ void __launch_bounds__(256) k_build_lists(
    const int* __restrict__ topk_id, const float* __restrict__ topk_w,
    int* __restrict__ counts, int* __restrict__ lt, int* __restrict__ ls,
    float* __restrict__ lw) {
  __shared__ int sc[256];
  __shared__ int base_s;
  int e = blockIdx.x, tid = threadIdx.x;
  if (tid == 0) base_s = 0;
  __syncthreads();

  for (int c = 0; c < Tn / 256; ++c) {
    int t = c * 256 + tid;
    int slot = -1;
    if (topk_id[2 * t] == e) slot = 0;
    else if (topk_id[2 * t + 1] == e) slot = 1;
    int cnt = (slot >= 0) ? 1 : 0;
    sc[tid] = cnt;
    __syncthreads();
    for (int off = 1; off < 256; off <<= 1) {           // inclusive scan
      int v = (tid >= off) ? sc[tid - off] : 0;
      __syncthreads();
      sc[tid] += v;
      __syncthreads();
    }
    int pos   = base_s + sc[tid] - cnt;
    int total = sc[255];
    __syncthreads();
    if (slot >= 0) {
      lt[e * Tn + pos] = t;
      ls[e * Tn + pos] = slot;
      lw[e * Tn + pos] = topk_w[2 * t + slot];
    }
    if (tid == 0) base_s += total;
    __syncthreads();
  }
  if (tid == 0) counts[e] = base_s;
}

// ------- WMMA fragment builders (documented CDNA5 16-bit layouts) ----------
// A 16x32: lane half hl selects K-half-of-8; VGPR j<4: K=2j+8*hl; j>=4: 16+2(j-4)+8*hl.
// Bytes form two 16B-aligned runs -> merges to 2x ds_load_b128.
__device__ __forceinline__ void load_fragA(FragU& f, const u16* row, int hl, int ks) {
#pragma unroll
  for (int j = 0; j < 8; ++j) {
    int kb = ks + ((j < 4) ? (2 * j) : (16 + 2 * (j - 4))) + hl * 8;
    f.u[j] = *(const u32*)(row + kb);
  }
}
// B 32x16 stored N-major: lane n = L&15; VGPR j<4: K=2j+8*hl; j>=4: 16+2(j-4)+8*hl.
// Bytes form two 16B-aligned runs -> 2x ds_load_b128.
__device__ __forceinline__ void load_fragB(FragU& f, const u16* row, int hl, int ks) {
#pragma unroll
  for (int j = 0; j < 8; ++j) {
    int kb = ks + ((j < 4) ? (2 * j) : (16 + 2 * (j - 4))) + hl * 8;
    f.u[j] = *(const u32*)(row + kb);
  }
}

// stage one BKxBN fp32 weight tile into N-major bf16 LDS.
// Each thread reads float4 along n for two adjacent k-rows (coalesced 128B
// bursts, 4 elements per VMEM op) and packs each (k,k+1) pair into one
// ds_store_b32.
template <int LDW>
__device__ __forceinline__ void stage_B(u16 (*Bs)[BK + APAD],
                                        const float* __restrict__ w,
                                        size_t wbase, int k0, int n0, int tid) {
  int n4 = (tid & 31) * 4;        // 32 threads cover 128 n in float4 chunks
  int kb = (tid >> 5) * 2;        // 8 thread-groups -> k pairs 0,2,...,14
#pragma unroll
  for (int i = 0; i < 4; ++i) {
    int k = kb + i * 16;          // covers all even k in [0,64)
    const float* src = w + wbase + (size_t)(k0 + k) * LDW + n0 + n4;
    f32x4 va = *(const f32x4*)src;          // row k
    f32x4 vb = *(const f32x4*)(src + LDW);  // row k+1
#pragma unroll
    for (int c = 0; c < 4; ++c) {
      *(u32*)(&Bs[n4 + c][k]) = (u32)f2bf(va[c]) | ((u32)f2bf(vb[c]) << 16);
    }
  }
}

// ---------------- kernel 4: expert stage-1  act = silu(X Wg) * (X Wu) ------
__global__ void __launch_bounds__(256) k_moe_gemm1(
    const u16* __restrict__ xb, const float* __restrict__ wg,
    const float* __restrict__ wu, const int* __restrict__ counts,
    const int* __restrict__ lt, const int* __restrict__ ls,
    u16* __restrict__ act) {
  int e   = blockIdx.z;
  int cnt = counts[e];
  int m0  = blockIdx.y * BM;
  if (m0 >= cnt) return;
  int n0  = blockIdx.x * BN;

  __shared__ u16 As[BM][BK + APAD];
  __shared__ u16 Bgs[BN][BK + APAD];
  __shared__ u16 Bus[BN][BK + APAD];
  __shared__ int tk_s[BM];
  __shared__ int ar_s[BM];

  int tid = threadIdx.x;
  if (tid < BM) {
    int r = m0 + tid;
    if (r < cnt) {
      int t = lt[e * Tn + r];
      tk_s[tid] = t;
      ar_s[tid] = 2 * t + ls[e * Tn + r];
    } else { tk_s[tid] = -1; ar_s[tid] = -1; }
  }
  __syncthreads();

  int wave = tid >> 5, lane = tid & 31;
  int hl = lane >> 4, ln = lane & 15;
  int mi = wave >> 2, ni = wave & 3;   // wave tile: rows mi*16, cols ni*32 (2 subtiles)

  // per-thread A staging coords: 8 threads/row, 16B each -> whole 32x64 tile
  int ra = tid >> 3;
  int kq = (tid & 7) * 8;

  v8f cg[2], cu[2];
#pragma unroll
  for (int s = 0; s < 2; ++s) {
    cg[s] = (v8f){0.f, 0.f, 0.f, 0.f, 0.f, 0.f, 0.f, 0.f};
    cu[s] = (v8f){0.f, 0.f, 0.f, 0.f, 0.f, 0.f, 0.f, 0.f};
  }

  const size_t wbase = (size_t)e * HD * ID;
  for (int k0 = 0; k0 < HD; k0 += BK) {
    {   // stage A: one 16B vector load/store per thread
      int t = tk_s[ra];
      u32x4 v = {0u, 0u, 0u, 0u};
      if (t >= 0) v = *(const u32x4*)(xb + (size_t)t * HD + k0 + kq);
      *(u32x4*)(&As[ra][kq]) = v;
    }
    stage_B<ID>(Bgs, wg, wbase, k0, n0, tid);
    stage_B<ID>(Bus, wu, wbase, k0, n0, tid);
    if (k0 + BK < HD) {   // cache hints for next chunk (2 lines per 512B row)
      int r = tid >> 1, half128 = (tid & 1) * 64;
      if (tid < 128)
        __builtin_prefetch(&wg[wbase + (size_t)(k0 + BK + r) * ID + n0 + half128], 0, 0);
      else
        __builtin_prefetch(&wu[wbase + (size_t)(k0 + BK + (r - 64)) * ID + n0 + half128], 0, 0);
    }
    __syncthreads();

#pragma unroll
    for (int ks = 0; ks < BK; ks += 32) {
      FragU a;
      load_fragA(a, &As[mi * 16 + ln][0], hl, ks);
#pragma unroll
      for (int s = 0; s < 2; ++s) {          // A fragment reused 4x
        FragU bg, bu;
        load_fragB(bg, &Bgs[ni * 32 + s * 16 + ln][0], hl, ks);
        load_fragB(bu, &Bus[ni * 32 + s * 16 + ln][0], hl, ks);
        cg[s] = __builtin_amdgcn_wmma_f32_16x16x32_bf16(false, a.v, false, bg.v, (short)0, cg[s], false, false);
        cu[s] = __builtin_amdgcn_wmma_f32_16x16x32_bf16(false, a.v, false, bu.v, (short)0, cu[s], false, false);
      }
    }
    __syncthreads();
  }

  // epilogue: silu(g)*u -> bf16 act rows (row = 2*token+slot)
#pragma unroll
  for (int s = 0; s < 2; ++s) {
#pragma unroll
    for (int r = 0; r < 8; ++r) {
      int mm = mi * 16 + hl * 8 + r;
      if (m0 + mm < cnt) {
        float g = cg[s][r], uu = cu[s][r];
        float sv = (g / (1.f + __expf(-g))) * uu;
        act[(size_t)ar_s[mm] * ID + n0 + ni * 32 + s * 16 + ln] = f2bf(sv);
      }
    }
  }
}

// ---------------- kernel 5: expert stage-2  out += w * (act Wd) ------------
__global__ void __launch_bounds__(256) k_moe_gemm2(
    const u16* __restrict__ act, const float* __restrict__ wd,
    const int* __restrict__ counts, const int* __restrict__ lt,
    const int* __restrict__ ls, const float* __restrict__ lw,
    float* __restrict__ out) {
  int e   = blockIdx.z;
  int cnt = counts[e];
  int m0  = blockIdx.y * BM;
  if (m0 >= cnt) return;
  int n0  = blockIdx.x * BN;

  __shared__ u16 As[BM][BK + APAD];
  __shared__ u16 Bds[BN][BK + APAD];
  __shared__ int   tk_s[BM];
  __shared__ int   ar_s[BM];
  __shared__ float wr_s[BM];

  int tid = threadIdx.x;
  if (tid < BM) {
    int r = m0 + tid;
    if (r < cnt) {
      int t = lt[e * Tn + r];
      tk_s[tid] = t;
      ar_s[tid] = 2 * t + ls[e * Tn + r];
      wr_s[tid] = lw[e * Tn + r];
    } else { tk_s[tid] = -1; ar_s[tid] = -1; wr_s[tid] = 0.f; }
  }
  __syncthreads();

  int wave = tid >> 5, lane = tid & 31;
  int hl = lane >> 4, ln = lane & 15;
  int mi = wave >> 2, ni = wave & 3;
  int ra = tid >> 3;
  int kq = (tid & 7) * 8;

  v8f c[2];
#pragma unroll
  for (int s = 0; s < 2; ++s)
    c[s] = (v8f){0.f, 0.f, 0.f, 0.f, 0.f, 0.f, 0.f, 0.f};

  const size_t wbase = (size_t)e * ID * HD;
  for (int k0 = 0; k0 < ID; k0 += BK) {
    {
      int ar = ar_s[ra];
      u32x4 v = {0u, 0u, 0u, 0u};
      if (ar >= 0) v = *(const u32x4*)(act + (size_t)ar * ID + k0 + kq);
      *(u32x4*)(&As[ra][kq]) = v;
    }
    stage_B<HD>(Bds, wd, wbase, k0, n0, tid);
    if (k0 + BK < ID && tid < 128) {
      int r = tid >> 1, half128 = (tid & 1) * 64;
      __builtin_prefetch(&wd[wbase + (size_t)(k0 + BK + r) * HD + n0 + half128], 0, 0);
    }
    __syncthreads();

#pragma unroll
    for (int ks = 0; ks < BK; ks += 32) {
      FragU a;
      load_fragA(a, &As[mi * 16 + ln][0], hl, ks);
#pragma unroll
      for (int s = 0; s < 2; ++s) {
        FragU b;
        load_fragB(b, &Bds[ni * 32 + s * 16 + ln][0], hl, ks);
        c[s] = __builtin_amdgcn_wmma_f32_16x16x32_bf16(false, a.v, false, b.v, (short)0, c[s], false, false);
      }
    }
    __syncthreads();
  }

  // epilogue: scale by routing weight, scatter-add (exactly 2 adds/element ->
  // fp-add commutativity keeps this deterministic)
#pragma unroll
  for (int s = 0; s < 2; ++s) {
#pragma unroll
    for (int r = 0; r < 8; ++r) {
      int mm = mi * 16 + hl * 8 + r;
      if (m0 + mm < cnt) {
        int t = tk_s[mm];
        atomicAdd(&out[(size_t)t * HD + n0 + ni * 32 + s * 16 + ln], c[s][r] * wr_s[mm]);
      }
    }
  }
}

// ---------------- host launch ----------------
extern "C" void kernel_launch(void* const* d_in, const int* in_sizes, int n_in,
                              void* d_out, int out_size, void* d_ws, size_t ws_size,
                              hipStream_t stream) {
  (void)in_sizes; (void)n_in; (void)out_size; (void)ws_size;
  const float* x  = (const float*)d_in[0];  // [T,H]
  const float* gw = (const float*)d_in[1];  // [E,H]
  const float* wg = (const float*)d_in[2];  // [E,H,I]
  const float* wu = (const float*)d_in[3];  // [E,H,I]
  const float* wd = (const float*)d_in[4];  // [E,I,H]
  float* out = (float*)d_out;               // [T,H] then [T,E] logits

  // workspace carve (≈27.6 MB)
  uintptr_t p = (uintptr_t)d_ws;
  auto carve = [&](size_t bytes) {
    p = (p + 255) & ~(uintptr_t)255;
    uintptr_t q = p; p += bytes; return (void*)q;
  };
  u16*   xb      = (u16*)  carve((size_t)Tn * HD * sizeof(u16));
  int*   topk_id = (int*)  carve((size_t)Tn * 2 * sizeof(int));
  float* topk_w  = (float*)carve((size_t)Tn * 2 * sizeof(float));
  int*   counts  = (int*)  carve((size_t)NE * sizeof(int));
  int*   lt      = (int*)  carve((size_t)NE * Tn * sizeof(int));
  int*   ls      = (int*)  carve((size_t)NE * Tn * sizeof(int));
  float* lw      = (float*)carve((size_t)NE * Tn * sizeof(float));
  u16*   act     = (u16*)  carve((size_t)2 * Tn * ID * sizeof(u16));

  float* logits = out + (size_t)Tn * HD;

  // out accumulates via atomics -> must be zeroed every call
  hipMemsetAsync(out, 0, (size_t)Tn * HD * sizeof(float), stream);

  k_cvt_bf16<<<(Tn * HD) / (4 * 256), 256, 0, stream>>>(x, xb);
  k_router<<<Tn / 8, 256, 0, stream>>>(x, gw, logits, topk_id, topk_w);
  k_build_lists<<<NE, 256, 0, stream>>>(topk_id, topk_w, counts, lt, ls, lw);
  k_moe_gemm1<<<dim3(ID / BN, Tn / BM, NE), 256, 0, stream>>>(
      xb, wg, wu, counts, lt, ls, act);
  k_moe_gemm2<<<dim3(HD / BN, Tn / BM, NE), 256, 0, stream>>>(
      act, wd, counts, lt, ls, lw, out);
}
